// LSTMSoftAttentionCoreZoneout_4312147165255
// MI455X (gfx1250) — matrix operand order repaired and checked by reading, not compiled, not run on hardware
//
#include <hip/hip_runtime.h>
#include <cmath>
#include <cstdint>

#define L_SEQ 196
#define BATCH 256
#define EDIM  1024
#define HDIM  1024
#define FDIM  2048
#define ADIM  512
#define MROWS (BATCH * L_SEQ)   // 50176, divisible by 32
#define G4H   4096              // 4*H

// ---------------- workspace layout (floats) ----------------
#define HL2_OFF  0u                       // [256*512]
#define SC_OFF   (HL2_OFF + 131072u)      // [50176]
#define WSM_OFF  (SC_OFF + 50176u)        // [50176]
#define Z_OFF    (WSM_OFF + 50176u)       // [256*2048]
#define SUM_OFF  (Z_OFF + 524288u)        // [256*4096]
#define CB_OFF   (SUM_OFF + 1048576u)     // [4096]
#define HB_OFF   (CB_OFF + 4096u)         // [512]

typedef __attribute__((ext_vector_type(16))) __bf16 v16bf;
typedef __attribute__((ext_vector_type(8)))  float  v8f;
typedef __attribute__((ext_vector_type(4)))  unsigned int v4u;
typedef __attribute__((ext_vector_type(8)))  int v8i;
typedef __attribute__((ext_vector_type(4)))  int v4i;

#define TSTR 33   // fp32 dwords per LDS tile row (32 data + 1 TDM pad dword)

#if __has_builtin(__builtin_amdgcn_tensor_load_to_lds) && \
    __has_builtin(__builtin_amdgcn_s_wait_tensorcnt)
#define HAS_TDM 1
#else
#define HAS_TDM 0
#endif

__device__ __forceinline__ v8f wmma_bf16(v16bf a, v16bf b, v8f c) {
  return __builtin_amdgcn_wmma_f32_16x16x32_bf16(false, a, false, b, (short)0, c,
                                                 false, false);
}

#if HAS_TDM
// DMA a [rows x 32] fp32 tile (global row stride = stride_elems) into LDS at
// lds_addr with 1 pad dword per 32-dword row (LDS row stride = 33 dwords).
// D# per CDNA5 ISA ch.8: group0 {count=1, lds_addr, global_addr, type=2},
// group1 {data_size=4B, pad_enable, pad_interval=32dw, pad_amount=1dw,
//         tensor_dim0=stride, tensor_dim1=rows, tile_dim0=32, tile_dim1=rows,
//         tensor_dim0_stride=stride}.
__device__ __forceinline__ void tdm_load_tile_f32(unsigned lds_addr,
                                                  const float* gptr,
                                                  unsigned rows,
                                                  unsigned stride_elems) {
  unsigned long long ga = (unsigned long long)(uintptr_t)gptr;
  v4u g0;
  g0[0] = 1u;                                            // count=1, user mode
  g0[1] = lds_addr;                                      // lds_addr [63:32]
  g0[2] = (unsigned)(ga & 0xFFFFFFFFu);                  // global_addr [95:64]
  g0[3] = (unsigned)((ga >> 32) & 0x01FFFFFFu) | (2u << 30);  // addr hi | type=2
  v8i g1;
  g1[0] = (int)((2u << 16) | (1u << 20) | (4u << 22));   // 4B elems, pad 1dw/32dw
  g1[1] = (int)((stride_elems & 0xFFFFu) << 16);         // tensor_dim0 lo16
  g1[2] = (int)(((stride_elems >> 16) & 0xFFFFu) | ((rows & 0xFFFFu) << 16));
  g1[3] = (int)(32u << 16);                              // dim1 hi16=0 | tile_dim0=32
  g1[4] = (int)(rows & 0xFFFFu);                         // tile_dim1 | tile_dim2=0
  g1[5] = (int)stride_elems;                             // tensor_dim0_stride lo32
  g1[6] = 0;
  g1[7] = 0;
  v4i z4 = {0, 0, 0, 0};
#if defined(__clang_major__) && (__clang_major__ >= 23)
  v8i z8 = {0, 0, 0, 0, 0, 0, 0, 0};
  __builtin_amdgcn_tensor_load_to_lds(g0, g1, z4, z4, z8, 0);
#else
  __builtin_amdgcn_tensor_load_to_lds(g0, g1, z4, z4, 0);
#endif
}
#endif

// Manual fallback staging: copy [rows x 32] fp32 tile into stride-33 LDS.
__device__ __forceinline__ void stage_tile_f32(float* dst, const float* src,
                                               int rows, int stride_elems,
                                               int tid) {
  int elems = rows * 32;
  for (int e = tid; e < elems; e += 256) {
    int r = e >> 5, k = e & 31;
    dst[r * TSTR + k] = src[(size_t)r * stride_elems + k];
  }
}

// A-fragment (16x32, fp32 LDS -> bf16). Lane t<16: row t, K=0..15;
// lane t>=16: row t-16, K=16..31.
__device__ __forceinline__ v16bf load_a_frag_f32(const float* s, int lane) {
  int row = lane & 15, hi = lane >> 4;
  v16bf a;
#pragma unroll
  for (int e = 0; e < 16; ++e) {
    int k = (hi << 4) + e;
    a[e] = (__bf16)s[row * TSTR + k];
  }
  return a;
}

// B-fragment (32x16, fp32 LDS -> bf16). Lane t<16: col t, K=0..15;
// lane t>=16: col t-16, K=16..31.
__device__ __forceinline__ v16bf load_b_frag_f32(const float* s, int lane) {
  int col = lane & 15, kb = (lane >> 4) << 4;
  v16bf b;
#pragma unroll
  for (int e = 0; e < 16; ++e) b[e] = (__bf16)s[col * TSTR + kb + e];
  return b;
}

// =====================================================================
// Generic GEMM: out[M,N] = A[M,K] * W[N,K]^T (+ bias[N])
// WG tile 32x128, 8 waves; wave = (rowTile 0..1, colGroup 0..3) ->
// 16 rows x 32 cols (2 WMMA N-tiles). Both tiles staged by the TDM.
// =====================================================================
__global__ __launch_bounds__(256) void gemm_nt_bf16(
    const float* __restrict__ A, const float* __restrict__ W,
    float* __restrict__ out, const float* __restrict__ bias,
    int M, int N, int K) {
  __shared__ float sA[32 * TSTR];
  __shared__ float sB[128 * TSTR];
  (void)M;
  const int tid = threadIdx.x, lane = tid & 31, wave = tid >> 5;
  const int rowTile = wave >> 2, colGroup = wave & 3;
  const int m0 = blockIdx.x * 32;
  const int n0 = blockIdx.y * 128;
#if HAS_TDM
  const unsigned sA_lds = (unsigned)(uintptr_t)(void*)&sA[0];
  const unsigned sB_lds = (unsigned)(uintptr_t)(void*)&sB[0];
#endif

  v8f acc0 = {0.f, 0.f, 0.f, 0.f, 0.f, 0.f, 0.f, 0.f};
  v8f acc1 = acc0;

  for (int k0 = 0; k0 < K; k0 += 32) {
#if HAS_TDM
    if (wave == 0) {
      tdm_load_tile_f32(sB_lds, W + (size_t)n0 * K + k0, 128u, (unsigned)K);
      tdm_load_tile_f32(sA_lds, A + (size_t)m0 * K + k0, 32u, (unsigned)K);
    }
    __builtin_amdgcn_s_wait_tensorcnt(0);
#else
    stage_tile_f32(sB, W + (size_t)n0 * K + k0, 128, K, tid);
    stage_tile_f32(sA, A + (size_t)m0 * K + k0, 32, K, tid);
#endif
    __syncthreads();
    v16bf a  = load_a_frag_f32(&sA[rowTile * 16 * TSTR], lane);
    v16bf b0 = load_b_frag_f32(&sB[(colGroup * 32 + 0)  * TSTR], lane);
    v16bf b1 = load_b_frag_f32(&sB[(colGroup * 32 + 16) * TSTR], lane);
    acc0 = wmma_bf16(a, b0, acc0);
    acc1 = wmma_bf16(a, b1, acc1);
    __syncthreads();
  }

  const int hi = lane >> 4, nc = lane & 15;
#pragma unroll
  for (int v = 0; v < 8; ++v) {
    int m  = m0 + rowTile * 16 + v + hi * 8;
    int nA = n0 + colGroup * 32 + nc;
    int nB = nA + 16;
    float r0 = acc0[v], r1 = acc1[v];
    if (bias) { r0 += bias[nA]; r1 += bias[nB]; }
    out[(size_t)m * N + nA] = r0;
    out[(size_t)m * N + nB] = r1;
  }
}

// =====================================================================
// Fused gate GEMM: sums[256,4096] = xt@Wi^T + h@Wh^T + z@Wz^T + cbias.
// Same tiling as gemm_nt_bf16; K-loop runs over the 3 segments into the
// same accumulators (one pass over `sums`, no RMW traffic).
// =====================================================================
__global__ __launch_bounds__(256) void gates_gemm_fused(
    const float* __restrict__ xt, const float* __restrict__ h,
    const float* __restrict__ z,  const float* __restrict__ Wi,
    const float* __restrict__ Wh, const float* __restrict__ Wz,
    const float* __restrict__ cbias, float* __restrict__ sums) {
  __shared__ float sA[32 * TSTR];
  __shared__ float sB[128 * TSTR];
  const int tid = threadIdx.x, lane = tid & 31, wave = tid >> 5;
  const int rowTile = wave >> 2, colGroup = wave & 3;
  const int m0 = blockIdx.x * 32;
  const int n0 = blockIdx.y * 128;
#if HAS_TDM
  const unsigned sA_lds = (unsigned)(uintptr_t)(void*)&sA[0];
  const unsigned sB_lds = (unsigned)(uintptr_t)(void*)&sB[0];
#endif

  v8f acc0 = {0.f, 0.f, 0.f, 0.f, 0.f, 0.f, 0.f, 0.f};
  v8f acc1 = acc0;

  for (int seg = 0; seg < 3; ++seg) {
    const float* A = (seg == 0) ? xt : (seg == 1) ? h : z;
    const float* W = (seg == 0) ? Wi : (seg == 1) ? Wh : Wz;
    const int    K = (seg == 2) ? FDIM : EDIM;
    for (int k0 = 0; k0 < K; k0 += 32) {
#if HAS_TDM
      if (wave == 0) {
        tdm_load_tile_f32(sB_lds, W + (size_t)n0 * K + k0, 128u, (unsigned)K);
        tdm_load_tile_f32(sA_lds, A + (size_t)m0 * K + k0, 32u, (unsigned)K);
      }
      __builtin_amdgcn_s_wait_tensorcnt(0);
#else
      stage_tile_f32(sB, W + (size_t)n0 * K + k0, 128, K, tid);
      stage_tile_f32(sA, A + (size_t)m0 * K + k0, 32, K, tid);
#endif
      __syncthreads();
      v16bf a  = load_a_frag_f32(&sA[rowTile * 16 * TSTR], lane);
      v16bf b0 = load_b_frag_f32(&sB[(colGroup * 32 + 0)  * TSTR], lane);
      v16bf b1 = load_b_frag_f32(&sB[(colGroup * 32 + 16) * TSTR], lane);
      acc0 = wmma_bf16(a, b0, acc0);
      acc1 = wmma_bf16(a, b1, acc1);
      __syncthreads();
    }
  }

  const int hi = lane >> 4, nc = lane & 15;
#pragma unroll
  for (int v = 0; v < 8; ++v) {
    int m  = m0 + rowTile * 16 + v + hi * 8;
    int nA = n0 + colGroup * 32 + nc;
    int nB = nA + 16;
    sums[(size_t)m * G4H + nA] = acc0[v] + cbias[nA];
    sums[(size_t)m * G4H + nB] = acc1[v] + cbias[nB];
  }
}

// =====================================================================
// Fused attention score: 32 rows x 256 A-cols per WG (grid.y = col half),
// WMMA over K=2048, epilogue tanh(acc + hl2[b,n]) * Wo[n], LDS row-reduce,
// atomicAdd into scores (pre-zeroed). bo dropped: softmax shift-invariant.
// Wave owns 16 rows x 64 cols (4 N-tiles). Both tiles staged via TDM.
// =====================================================================
__global__ __launch_bounds__(256) void attn_score_kernel(
    const float* __restrict__ att,   // [MROWS, FDIM]
    const float* __restrict__ Wa,    // [ADIM, FDIM]
    const float* __restrict__ hl2,   // [BATCH, ADIM]
    const float* __restrict__ Wo,    // [ADIM]
    float* __restrict__ scores) {    // [MROWS]
  __shared__ float sA[32 * TSTR];
  __shared__ float sB[256 * TSTR];
  __shared__ float sScore[32];
  const int tid = threadIdx.x, lane = tid & 31, wave = tid >> 5;
  const int rowTile = wave >> 2, colGroup = wave & 3;
  const int m0 = blockIdx.x * 32;
  const int n0 = blockIdx.y * 256;
#if HAS_TDM
  const unsigned sA_lds = (unsigned)(uintptr_t)(void*)&sA[0];
  const unsigned sB_lds = (unsigned)(uintptr_t)(void*)&sB[0];
#endif

  v8f acc[4];
  v8f zero = {0.f, 0.f, 0.f, 0.f, 0.f, 0.f, 0.f, 0.f};
#pragma unroll
  for (int t = 0; t < 4; ++t) acc[t] = zero;

  for (int k0 = 0; k0 < FDIM; k0 += 32) {
#if HAS_TDM
    if (wave == 0) {
      tdm_load_tile_f32(sB_lds, Wa + (size_t)n0 * FDIM + k0, 256u, FDIM);
      tdm_load_tile_f32(sA_lds, att + (size_t)m0 * FDIM + k0, 32u, FDIM);
    }
    if (k0 + 32 < FDIM)   // warm GL2 for the next A slice
      __builtin_prefetch(att + (size_t)(m0 + (tid >> 4)) * FDIM + k0 + 32 +
                             ((tid & 15) << 1), 0, 1);
    __builtin_amdgcn_s_wait_tensorcnt(0);
#else
    stage_tile_f32(sB, Wa + (size_t)n0 * FDIM + k0, 256, FDIM, tid);
    stage_tile_f32(sA, att + (size_t)m0 * FDIM + k0, 32, FDIM, tid);
#endif
    __syncthreads();
    v16bf a = load_a_frag_f32(&sA[rowTile * 16 * TSTR], lane);
#pragma unroll
    for (int t = 0; t < 4; ++t) {
      v16bf b = load_b_frag_f32(&sB[(colGroup * 64 + t * 16) * TSTR], lane);
      acc[t] = wmma_bf16(a, b, acc[t]);
    }
    __syncthreads();
  }

  if (tid < 32) sScore[tid] = 0.f;
  __syncthreads();

  const int hi = lane >> 4, nc = lane & 15;
#pragma unroll
  for (int v = 0; v < 8; ++v) {
    int mLoc = rowTile * 16 + v + hi * 8;
    int gRow = m0 + mLoc;
    int b    = gRow / L_SEQ;
    float part = 0.f;
#pragma unroll
    for (int t = 0; t < 4; ++t) {
      int n = n0 + colGroup * 64 + t * 16 + nc;
      float x = acc[t][v] + hl2[(size_t)b * ADIM + n];
      part += tanhf(x) * Wo[n];
    }
    atomicAdd(&sScore[mLoc], part);
  }
  __syncthreads();
  if (tid < 32) atomicAdd(&scores[m0 + tid], sScore[tid]);
}

// ---------------- softmax over L=196 per batch row ----------------
__global__ __launch_bounds__(256) void softmax196(const float* __restrict__ scores,
                                                  float* __restrict__ w) {
  __shared__ float buf[256];
  const int b = blockIdx.x, t = threadIdx.x;
  float v = (t < L_SEQ) ? scores[b * L_SEQ + t] : -3.4e38f;
  buf[t] = v; __syncthreads();
  for (int s = 128; s > 0; s >>= 1) {
    if (t < s) buf[t] = fmaxf(buf[t], buf[t + s]);
    __syncthreads();
  }
  float mx = buf[0]; __syncthreads();
  float e = (t < L_SEQ) ? expf(v - mx) : 0.f;
  buf[t] = e; __syncthreads();
  for (int s = 128; s > 0; s >>= 1) {
    if (t < s) buf[t] += buf[t + s];
    __syncthreads();
  }
  float inv = 1.f / buf[0];
  if (t < L_SEQ) w[b * L_SEQ + t] = e * inv;
}

// ---------------- context: z[b,f] = sum_l w[b,l] * att[b,l,f] ----------------
__global__ __launch_bounds__(256) void context_kernel(
    const float* __restrict__ att, const float* __restrict__ w,
    float* __restrict__ z) {
  const int f = blockIdx.x * 256 + threadIdx.x;
  const int b = blockIdx.y;
  const float* wp = w + b * L_SEQ;
  const float* ap = att + (size_t)b * L_SEQ * FDIM + f;
  float acc = 0.f;
  for (int l = 0; l < L_SEQ; ++l) acc += wp[l] * ap[(size_t)l * FDIM];
  z[(size_t)b * FDIM + f] = acc;
}

// ---------------- gates + zoneout ----------------
__global__ __launch_bounds__(256) void gates_zoneout(
    const float* __restrict__ sums, const float* __restrict__ h,
    const float* __restrict__ c, float* __restrict__ out, int BH) {
  int i = blockIdx.x * 256 + threadIdx.x;
  if (i >= BH) return;
  int b = i >> 10, hh = i & 1023;
  const float* row = sums + (size_t)b * G4H;
  float ig = 1.f / (1.f + expf(-row[hh]));
  float fg = 1.f / (1.f + expf(-row[1024 + hh]));
  float og = 1.f / (1.f + expf(-row[2048 + hh]));
  float g  = tanhf(row[3072 + hh]);
  float cv = c[i], hv = h[i];
  float nc = cv * 0.5f + 0.5f * (fg * cv + ig * g);
  float nh = hv * 0.5f + 0.5f * (og * tanhf(nc));
  out[i] = nh;
  out[BH + i] = nh;
  out[2 * BH + i] = nc;
}

// ---------------- small utility kernels ----------------
__global__ void add3_kernel(const float* __restrict__ a, const float* __restrict__ b,
                            const float* __restrict__ c, float* __restrict__ o, int n) {
  int i = blockIdx.x * 256 + threadIdx.x;
  if (i < n) o[i] = a[i] + b[i] + (c ? c[i] : 0.f);
}
__global__ void zero_kernel(float* __restrict__ p, int n) {
  int i = blockIdx.x * 256 + threadIdx.x;
  if (i < n) p[i] = 0.f;
}

// =====================================================================
extern "C" void kernel_launch(void* const* d_in, const int* in_sizes, int n_in,
                              void* d_out, int out_size, void* d_ws, size_t ws_size,
                              hipStream_t stream) {
  (void)in_sizes; (void)n_in; (void)out_size; (void)ws_size;
  const float* xt  = (const float*)d_in[0];
  const float* att = (const float*)d_in[1];
  const float* h   = (const float*)d_in[2];
  const float* c   = (const float*)d_in[3];
  const float* Wi  = (const float*)d_in[4];
  const float* bi  = (const float*)d_in[5];
  const float* Wh  = (const float*)d_in[6];
  const float* bh  = (const float*)d_in[7];
  const float* Wz  = (const float*)d_in[8];
  const float* bz  = (const float*)d_in[9];
  const float* Wa  = (const float*)d_in[10];
  const float* ba  = (const float*)d_in[11];
  const float* Wha = (const float*)d_in[12];
  const float* bha = (const float*)d_in[13];
  const float* Wo  = (const float*)d_in[14];
  // d_in[15] = bo: softmax shift-invariant -> unused.

  float* ws     = (float*)d_ws;
  float* hl2    = ws + HL2_OFF;
  float* scores = ws + SC_OFF;
  float* wsm    = ws + WSM_OFF;
  float* zbuf   = ws + Z_OFF;
  float* sums   = ws + SUM_OFF;
  float* cbias  = ws + CB_OFF;
  float* hbias  = ws + HB_OFF;
  float* out    = (float*)d_out;

  add3_kernel<<<(ADIM + 255) / 256, 256, 0, stream>>>(bha, ba, nullptr, hbias, ADIM);
  add3_kernel<<<(G4H + 255) / 256, 256, 0, stream>>>(bi, bh, bz, cbias, G4H);
  zero_kernel<<<(MROWS + 255) / 256, 256, 0, stream>>>(scores, MROWS);

  // hl2 = h @ Wha^T + (bha + ba)
  gemm_nt_bf16<<<dim3(BATCH / 32, ADIM / 128), 256, 0, stream>>>(
      h, Wha, hl2, hbias, BATCH, ADIM, HDIM);

  // attention scores (dominant GEMM, fused tanh/dot epilogue)
  attn_score_kernel<<<dim3(MROWS / 32, ADIM / 256), 256, 0, stream>>>(
      att, Wa, hl2, Wo, scores);

  softmax196<<<BATCH, 256, 0, stream>>>(scores, wsm);

  context_kernel<<<dim3(FDIM / 256, BATCH), 256, 0, stream>>>(att, wsm, zbuf);

  // sums = xt@Wi^T + h@Wh^T + z@Wz^T + cbias (single fused pass)
  gates_gemm_fused<<<dim3(BATCH / 32, G4H / 128), 256, 0, stream>>>(
      xt, h, zbuf, Wi, Wh, Wz, cbias, sums);

  gates_zoneout<<<(BATCH * HDIM + 255) / 256, 256, 0, stream>>>(
      sums, h, c, out, BATCH * HDIM);
}
